// complexLSTM_56221121905335
// MI455X (gfx1250) — matrix operand order repaired and checked by reading, not compiled, not run on hardware
//
#include <hip/hip_runtime.h>
#include <hip/hip_bf16.h>

// Problem constants (match reference)
#define T_LEN  512
#define BATCH  64
#define HID    512
#define G4     2048            // 4*HID
#define GOFS   (512 * HID)     // element offset between gate blocks in W rows
#define NLAYER 2
// 16 blocks * 256 threads = 128 wave32 waves; wave w owns m-tile (w&3) and
// j-tile (w>>2) and computes all 4 gates x {r,i} for that 16x16 patch.
#define NB 16
#define NT 256

typedef __attribute__((ext_vector_type(16))) __bf16   v16bf;
typedef __attribute__((ext_vector_type(8)))  __bf16   v8bf;
typedef __attribute__((ext_vector_type(8)))  float    v8f;

static __device__ __forceinline__ v16bf bfcat(v8bf l, v8bf h) {
  return __builtin_shufflevector(l, h, 0,1,2,3,4,5,6,7,8,9,10,11,12,13,14,15);
}

// A-fragment (16x32 bf16) from f32 row-major memory; p = &row[k0 + (lane>>4)*8].
static __device__ __forceinline__ v16bf ldA_f32(const float* p) {
  const float4* q = (const float4*)p;
  float4 a = q[0], b = q[1], c = q[4], d = q[5];
  v16bf r;
  r[0]=(__bf16)a.x;  r[1]=(__bf16)a.y;  r[2]=(__bf16)a.z;  r[3]=(__bf16)a.w;
  r[4]=(__bf16)b.x;  r[5]=(__bf16)b.y;  r[6]=(__bf16)b.z;  r[7]=(__bf16)b.w;
  r[8]=(__bf16)c.x;  r[9]=(__bf16)c.y;  r[10]=(__bf16)c.z; r[11]=(__bf16)c.w;
  r[12]=(__bf16)d.x; r[13]=(__bf16)d.y; r[14]=(__bf16)d.z; r[15]=(__bf16)d.w;
  return r;
}
// A-fragment from bf16 memory (8 elems at p, 8 at p+16).
static __device__ __forceinline__ v16bf ldA_bf(const __bf16* p) {
  v8bf l = __builtin_bit_cast(v8bf, *(const uint4*)p);
  v8bf h = __builtin_bit_cast(v8bf, *(const uint4*)(p + 16));
  return bfcat(l, h);
}
// B-fragment (32x16, B = W^T): 8 bf16 at p, 8 at p+16; p = &Wrow[k0 + (lane>>4)*8].
static __device__ __forceinline__ v16bf ldB_bf(const __bf16* p) {
  v8bf l = __builtin_bit_cast(v8bf, *(const uint4*)p);
  v8bf h = __builtin_bit_cast(v8bf, *(const uint4*)(p + 16));
  return bfcat(l, h);
}

#define WMMA_BF16(a,b,c) \
  __builtin_amdgcn_wmma_f32_16x16x32_bf16(false,(a),false,(b),(short)0,(c),false,false)

static __device__ __forceinline__ float sigm(float x)  { return 1.f / (1.f + __expf(-x)); }
static __device__ __forceinline__ float tanh_(float x) { float e = __expf(2.f*x); return 1.f - 2.f/(e + 1.f); }

// Monotonic counting grid barrier (agent scope); `bar` memset to 0 per launch.
static __device__ __forceinline__ void gridbar(unsigned* bar, unsigned& round) {
  __threadfence();
  __syncthreads();
  round += 1;
  if (threadIdx.x == 0) {
    __hip_atomic_fetch_add(bar, 1u, __ATOMIC_RELEASE, __HIP_MEMORY_SCOPE_AGENT);
    const unsigned target = round * (unsigned)gridDim.x;
    while (__hip_atomic_load(bar, __ATOMIC_ACQUIRE, __HIP_MEMORY_SCOPE_AGENT) < target)
      __builtin_amdgcn_s_sleep(1);
  }
  __syncthreads();
  __threadfence();
}

template<bool NEG>
static __device__ __forceinline__ void cvt_arr(const float* s, __bf16* d, size_t n,
                                               unsigned tid, unsigned stride) {
  const float4* q = (const float4*)s;
  const size_t n4 = n >> 2;
  for (size_t i = tid; i < n4; i += stride) {
    float4 v = q[i];
    size_t o = i * 4;
    if (NEG) {
      d[o]=(__bf16)(-v.x); d[o+1]=(__bf16)(-v.y); d[o+2]=(__bf16)(-v.z); d[o+3]=(__bf16)(-v.w);
    } else {
      d[o]=(__bf16)(v.x);  d[o+1]=(__bf16)(v.y);  d[o+2]=(__bf16)(v.z);  d[o+3]=(__bf16)(v.w);
    }
  }
}

// One gate: 6 B-fragment loads (incl. pre-negated Wii/Whi) + 8 WMMAs.
#define DO_GATE(accR, accI, OFS) do {                         \
    v16bf b_ir  = ldB_bf(wir_p  + (OFS));                     \
    v16bf b_ii  = ldB_bf(wii_p  + (OFS));                     \
    v16bf b_iin = ldB_bf(wiin_p + (OFS));                     \
    v16bf b_hr  = ldB_bf(whr_p  + (OFS));                     \
    v16bf b_hi  = ldB_bf(whi_p  + (OFS));                     \
    v16bf b_hin = ldB_bf(whin_p + (OFS));                     \
    accR = WMMA_BF16(axr, b_ir,  accR);                       \
    accR = WMMA_BF16(axi, b_iin, accR);                       \
    accR = WMMA_BF16(ahr, b_hr,  accR);                       \
    accR = WMMA_BF16(ahi, b_hin, accR);                       \
    accI = WMMA_BF16(axi, b_ir,  accI);                       \
    accI = WMMA_BF16(axr, b_ii,  accI);                       \
    accI = WMMA_BF16(ahi, b_hr,  accI);                       \
    accI = WMMA_BF16(ahr, b_hi,  accI);                       \
  } while (0)

template<bool XBF>
__global__ void __launch_bounds__(NT, 1)
clstm_layer(const float* Xr32, const float* Xi32,    // f32 X: GEMM src (!XBF) / cvt src (XBF&&cvt_x)
            __bf16* xbr, __bf16* xbi,                // bf16 X buffers (GEMM src when XBF)
            __bf16* ybr, __bf16* ybi,                // bf16 Y (next layer's X); nullptr = skip
            float* Yr, float* Yi,                    // f32 Y -> d_out
            const float* Wir, const float* Wii, const float* Whr, const float* Whi,
            const float* bir, const float* bii, const float* bhr, const float* bhi,
            const float* h0r, const float* h0i, const float* c0r, const float* c0i,
            float* finr, float* fini,
            __bf16* wbir, __bf16* wbii, __bf16* wbiin,
            __bf16* wbhr, __bf16* wbhi, __bf16* wbhin,
            __bf16* hbr, __bf16* hbi, float* cre, float* cim,
            unsigned* bar, int cvt_x, int two_phase)
{
  const unsigned tid    = blockIdx.x * blockDim.x + threadIdx.x;
  const unsigned stride = gridDim.x * blockDim.x;
  const size_t   BH     = (size_t)BATCH * HID;
  const size_t   WN     = (size_t)G4 * HID;
  unsigned round = 0;

  // ---- pre-phase: bf16 weight copies (+negated Wii/Whi), h/c init, opt. X cvt
  cvt_arr<false>(Wir, wbir,  WN, tid, stride);
  cvt_arr<false>(Wii, wbii,  WN, tid, stride);
  cvt_arr<true >(Wii, wbiin, WN, tid, stride);
  cvt_arr<false>(Whr, wbhr,  WN, tid, stride);
  cvt_arr<false>(Whi, wbhi,  WN, tid, stride);
  cvt_arr<true >(Whi, wbhin, WN, tid, stride);
  if (XBF && cvt_x) {
    const size_t TBH = (size_t)T_LEN * BH;
    cvt_arr<false>(Xr32, xbr, TBH, tid, stride);
    cvt_arr<false>(Xi32, xbi, TBH, tid, stride);
  }
  for (unsigned e = tid; e < (unsigned)(BATCH * HID); e += stride) {
    hbr[e] = (__bf16)h0r[e];
    hbi[e] = (__bf16)h0i[e];
    cre[e] = c0r[e];
    cim[e] = c0i[e];
  }
  gridbar(bar, round);

  // Wave tile assignment
  const unsigned wave = tid >> 5;            // 0..127
  const unsigned lane = threadIdx.x & 31;
  const unsigned lr   = lane & 15;
  const unsigned lh   = lane >> 4;
  const unsigned m0   = (wave & 3) * 16;     // batch-row tile
  const unsigned j0   = wave >> 2;           // 0..31 hidden-column tile
  const unsigned j    = j0 * 16 + lr;        // this lane's column / W-row base

  // Per-gate fused biases (bi + bh), same for all 8 acc elements of a lane
  const float bR0 = bir[j]           + bhr[j];
  const float bR1 = bir[j +   HID]   + bhr[j +   HID];
  const float bR2 = bir[j + 2*HID]   + bhr[j + 2*HID];
  const float bR3 = bir[j + 3*HID]   + bhr[j + 3*HID];
  const float bI0 = bii[j]           + bhi[j];
  const float bI1 = bii[j +   HID]   + bhi[j +   HID];
  const float bI2 = bii[j + 2*HID]   + bhi[j + 2*HID];
  const float bI3 = bii[j + 3*HID]   + bhi[j + 3*HID];

  // B-fragment row base pointers (gate selected via constant GOFS offsets)
  const __bf16* wir_p  = wbir  + (size_t)j * HID;
  const __bf16* wii_p  = wbii  + (size_t)j * HID;
  const __bf16* wiin_p = wbiin + (size_t)j * HID;
  const __bf16* whr_p  = wbhr  + (size_t)j * HID;
  const __bf16* whi_p  = wbhi  + (size_t)j * HID;
  const __bf16* whin_p = wbhin + (size_t)j * HID;

  const __bf16* hr_row = hbr + (size_t)(m0 + lr) * HID;
  const __bf16* hi_row = hbi + (size_t)(m0 + lr) * HID;

  for (int t = 0; t < T_LEN; ++t) {
    // ---------- GEMM: all four gates, r & i, fused input+hidden projections --
    v8f aR0, aR1, aR2, aR3, aI0, aI1, aI2, aI3;
    #pragma unroll
    for (int v = 0; v < 8; ++v) {
      aR0[v]=bR0; aR1[v]=bR1; aR2[v]=bR2; aR3[v]=bR3;
      aI0[v]=bI0; aI1[v]=bI1; aI2[v]=bI2; aI3[v]=bI3;
    }
    const float*  xr32_row = Xr32 + (size_t)t*BH + (size_t)(m0+lr)*HID;
    const float*  xi32_row = Xi32 + (size_t)t*BH + (size_t)(m0+lr)*HID;
    const __bf16* xbr_row  = xbr  + (size_t)t*BH + (size_t)(m0+lr)*HID;
    const __bf16* xbi_row  = xbi  + (size_t)t*BH + (size_t)(m0+lr)*HID;

    for (int k0 = 0; k0 < HID; k0 += 32) {
      const int ka = k0 + (int)lh * 8;
      const int kb = k0 + (int)lh * 8;
      v16bf axr, axi;
      if constexpr (XBF) { axr = ldA_bf (xbr_row  + ka); axi = ldA_bf (xbi_row  + ka); }
      else               { axr = ldA_f32(xr32_row + ka); axi = ldA_f32(xi32_row + ka); }
      v16bf ahr = ldA_bf(hr_row + ka);
      v16bf ahi = ldA_bf(hi_row + ka);
      DO_GATE(aR0, aI0, kb + 0*GOFS);   // i gate
      DO_GATE(aR1, aI1, kb + 1*GOFS);   // f gate
      DO_GATE(aR2, aI2, kb + 2*GOFS);   // g gate
      DO_GATE(aR3, aI3, kb + 3*GOFS);   // o gate
    }

    if (two_phase) gridbar(bar, round);  // only when X/Y alias (plan A layer 1)

    // ---------- elementwise: gates + cell/hidden update, in registers --------
    {
      const size_t ebase = (size_t)(m0 + lh*8) * HID + j;
      float*  yr_t  = Yr + (size_t)t * BH;
      float*  yi_t  = Yi + (size_t)t * BH;
      __bf16* ybr_t = ybr + (size_t)t * BH;   // only dereferenced if ybr != null
      __bf16* ybi_t = ybi + (size_t)t * BH;
      const bool wfin = (t == T_LEN - 1);
      #pragma unroll
      for (int v = 0; v < 8; ++v) {
        const size_t e = ebase + (size_t)v * HID;
        float i_r = sigm (aR0[v]), f_r = sigm (aR1[v]);
        float g_r = tanh_(aR2[v]), o_r = sigm (aR3[v]);
        float i_i = sigm (aI0[v]), f_i = sigm (aI1[v]);
        float g_i = tanh_(aI2[v]), o_i = sigm (aI3[v]);
        float crv = cre[e], civ = cim[e];
        float cr2 = f_r*crv - f_i*civ + i_r*g_r - i_i*g_i;
        float ci2 = f_r*civ + f_i*crv + i_r*g_i + i_i*g_r;
        float tr  = tanh_(cr2), ti = tanh_(ci2);
        float hr2 = o_r*tr - o_i*ti;
        float hi2 = o_r*ti + o_i*tr;
        cre[e] = cr2; cim[e] = ci2;
        hbr[e] = (__bf16)hr2;
        hbi[e] = (__bf16)hi2;
        yr_t[e] = hr2; yi_t[e] = hi2;
        if (ybr) { ybr_t[e] = (__bf16)hr2; ybi_t[e] = (__bf16)hi2; }
        if (wfin) { finr[e] = hr2; fini[e] = hi2; }
      }
    }
    gridbar(bar, round);
  }
}

extern "C" void kernel_launch(void* const* d_in, const int* in_sizes, int n_in,
                              void* d_out, int out_size, void* d_ws, size_t ws_size,
                              hipStream_t stream) {
  (void)in_sizes; (void)n_in; (void)out_size;
  const float* seq_r = (const float*)d_in[0];
  const float* seq_i = (const float*)d_in[1];
  const float* h0r   = (const float*)d_in[2];
  const float* h0i   = (const float*)d_in[3];
  const float* c0r   = (const float*)d_in[4];
  const float* c0i   = (const float*)d_in[5];
  const float* Wir   = (const float*)d_in[6];
  const float* Wii   = (const float*)d_in[7];
  const float* Whr   = (const float*)d_in[8];
  const float* Whi   = (const float*)d_in[9];
  const float* bir   = (const float*)d_in[10];
  const float* bii   = (const float*)d_in[11];
  const float* bhr   = (const float*)d_in[12];
  const float* bhi   = (const float*)d_in[13];

  float* out = (float*)d_out;
  const size_t TBH = (size_t)T_LEN * BATCH * HID;
  const size_t BH  = (size_t)BATCH * HID;
  const size_t WN  = (size_t)G4 * HID;
  float* yr   = out;
  float* yi   = out + TBH;
  float* finr = out + 2 * TBH;
  float* fini = finr + (size_t)NLAYER * BH;

  // Workspace carve
  char* w = (char*)d_ws;
  unsigned* bar = (unsigned*)w;                       // 256 B (memset per layer)
  __bf16* wbir  = (__bf16*)(w + 256);
  __bf16* wbii  = wbir  + WN;
  __bf16* wbiin = wbii  + WN;
  __bf16* wbhr  = wbiin + WN;
  __bf16* wbhi  = wbhr  + WN;
  __bf16* wbhin = wbhi  + WN;
  __bf16* hbr   = wbhin + WN;
  __bf16* hbi   = hbr + BH;
  float*  cre   = (float*)(hbi + BH);
  float*  cim   = cre + BH;
  __bf16* xAr   = (__bf16*)(cim + BH);                // plan B only beyond here
  __bf16* xAi   = xAr + TBH;
  __bf16* xBr   = xAi + TBH;
  __bf16* xBi   = xBr + TBH;

  const size_t base_need  = 256 + 6*WN*sizeof(__bf16) + 2*BH*sizeof(__bf16) + 2*BH*sizeof(float);
  const size_t planB_need = base_need + 4*TBH*sizeof(__bf16);   // ~147 MB
  const bool   planB      = (ws_size >= planB_need);

  for (int l = 0; l < NLAYER; ++l) {
    hipMemsetAsync(bar, 0, 256, stream);
    const float* Wir_l = Wir + (size_t)l*WN, *Wii_l = Wii + (size_t)l*WN;
    const float* Whr_l = Whr + (size_t)l*WN, *Whi_l = Whi + (size_t)l*WN;
    const float* bir_l = bir + (size_t)l*G4, *bii_l = bii + (size_t)l*G4;
    const float* bhr_l = bhr + (size_t)l*G4, *bhi_l = bhi + (size_t)l*G4;
    const float* h0r_l = h0r + (size_t)l*BH, *h0i_l = h0i + (size_t)l*BH;
    const float* c0r_l = c0r + (size_t)l*BH, *c0i_l = c0i + (size_t)l*BH;
    float* finr_l = finr + (size_t)l*BH, *fini_l = fini + (size_t)l*BH;

    if (planB) {
      // bf16 sequence path: layer0 cvt seq->xA, writes xB; layer1 reads xB.
      __bf16* xr_in  = (l == 0) ? xAr : xBr;
      __bf16* xi_in  = (l == 0) ? xAi : xBi;
      __bf16* yr_bf  = (l == 0) ? xBr : nullptr;
      __bf16* yi_bf  = (l == 0) ? xBi : nullptr;
      clstm_layer<true><<<NB, NT, 0, stream>>>(
          seq_r, seq_i, xr_in, xi_in, yr_bf, yi_bf, yr, yi,
          Wir_l, Wii_l, Whr_l, Whi_l, bir_l, bii_l, bhr_l, bhi_l,
          h0r_l, h0i_l, c0r_l, c0i_l, finr_l, fini_l,
          wbir, wbii, wbiin, wbhr, wbhi, wbhin, hbr, hbi, cre, cim,
          bar, /*cvt_x=*/(l == 0) ? 1 : 0, /*two_phase=*/0);
    } else {
      // f32 sequence path; layer1 runs in place on d_out -> needs 2nd barrier.
      const float* Xr = (l == 0) ? seq_r : yr;
      const float* Xi = (l == 0) ? seq_i : yi;
      clstm_layer<false><<<NB, NT, 0, stream>>>(
          Xr, Xi, nullptr, nullptr, nullptr, nullptr, yr, yi,
          Wir_l, Wii_l, Whr_l, Whi_l, bir_l, bii_l, bhr_l, bhi_l,
          h0r_l, h0i_l, c0r_l, c0i_l, finr_l, fini_l,
          wbir, wbii, wbiin, wbhr, wbhi, wbhin, hbr, hbi, cre, cim,
          bar, /*cvt_x=*/0, /*two_phase=*/(l == 0) ? 0 : 1);
    }
  }
}
